// RNNEncoder_38783554683288
// MI455X (gfx1250) — matrix operand order repaired and checked by reading, not compiled, not run on hardware
//
#include <hip/hip_runtime.h>
#include <hip/hip_bf16.h>

typedef __attribute__((ext_vector_type(16))) __bf16 v16bf;
typedef __attribute__((ext_vector_type(8)))  float  v8f;

#define Bc 64
#define Tc 12
#define Fc 64
#define Nc 1024
#define Hc 256
#define Gc 768      // 3*H
#define SMEM_BYTES 262144   // 4 teams * 2 layers * pingpong * 2 ntiles * 8KB

union Frag16 { v16bf v; unsigned short us[16]; uint4 q[2]; };
union Chunk8 { uint4 q; unsigned short us[8]; };

__device__ __forceinline__ unsigned short f2bf(float f) {
  unsigned int u = __float_as_uint(f);
  u += 0x7FFFu + ((u >> 16) & 1u);            // round-to-nearest-even
  return (unsigned short)(u >> 16);
}
__device__ __forceinline__ float bf2f(unsigned short h) {
  return __uint_as_float(((unsigned int)h) << 16);
}
__device__ __forceinline__ float sigm(float x) { return 1.0f / (1.0f + __expf(-x)); }

// ---------------------------------------------------------------------------
// Pack a (K x 3H) row-major fp32 weight matrix into bf16 A-operand fragments
// in a contiguous-k 16x32 A layout (same k order as the B operand):
//   lane l: M = l&15, khalf = l>>4
//   element j (0..15): k_local = khalf*16 + j
// Fragment index = (mtile*KT + kt); each fragment = 32 lanes * 16 bf16.
// ---------------------------------------------------------------------------
__global__ void pack_weights(const float* __restrict__ W,
                             unsigned short* __restrict__ out, int KT) {
  int idx = blockIdx.x * blockDim.x + threadIdx.x;
  int total = (Gc / 16) * KT * 512;
  if (idx >= total) return;
  int j     = idx & 15;
  int lane  = (idx >> 4) & 31;
  int frag  = idx >> 9;
  int kt    = frag % KT;
  int mtile = frag / KT;
  int g     = mtile * 16 + (lane & 15);
  int khalf = lane >> 4;
  int k     = kt * 32 + khalf * 16 + j;
  out[idx] = f2bf(W[(size_t)k * Gc + g]);
}

// ---------------------------------------------------------------------------
// Fragment helpers
// ---------------------------------------------------------------------------
__device__ __forceinline__ void wmma_bf16(v8f& acc, const Frag16& a, const Frag16& b) {
  acc = __builtin_amdgcn_wmma_f32_16x16x32_bf16(
      /*neg_a=*/false, a.v, /*neg_b=*/false, b.v,
      /*c_mod=*/(short)0, acc, /*reuse_a=*/false, /*reuse_b=*/false);
}

__device__ __forceinline__ Frag16 load_afrag(const unsigned short* __restrict__ p,
                                             int mtile, int KT, int kt, int lane) {
  Frag16 f;
  const uint4* q = (const uint4*)(p + (size_t)((mtile * KT + kt) * 512 + lane * 16));
  f.q[0] = q[0]; f.q[1] = q[1];
  return f;
}

// B operand (K x 16) bf16 layout in LDS: element j of lane l maps to
// k = kt*32 + (l>>4)*16 + j, n = l&15 -> stored at buf[kt*512 + l*16 + j].
// buf stays a plain arithmetic expression rooted at the extern __shared__
// symbol so infer-address-spaces emits ds_load_b128 (not flat).
__device__ __forceinline__ Frag16 load_bfrag(const unsigned short* buf, int kt, int lane) {
  Frag16 f;
  const uint4* q = (const uint4*)(buf + kt * 512 + lane * 16);
  f.q[0] = q[0]; f.q[1] = q[1];
  return f;
}

// ---------------------------------------------------------------------------
// GRU pointwise update for one 16x16 gate-row tile / one sensor tile.
//   h' = (1-z)*n + z*h,  n = tanh(nx + bn + r*nh)
// D layout: element r -> h row m = ht*16 + khalf*8 + r, column n; the 8 rows
// are contiguous -> one b128 chunk in the B-layout state buffer.
// ---------------------------------------------------------------------------
__device__ __forceinline__ void gru_update(
    const v8f& ar, const v8f& az, const v8f& anx, const v8f& anh,
    const float* __restrict__ bias,
    const unsigned short* hcur, unsigned short* hnxt,
    float* outp, int ht, int lane)
{
  const int n     = lane & 15;
  const int khalf = lane >> 4;
  const int m0  = ht * 16 + khalf * 8;
  const int off = (ht >> 1) * 512 + (n + ((ht & 1) << 4)) * 16 + khalf * 8;
  Chunk8 hold, hnew;
  hold.q = *(const uint4*)(hcur + off);
#pragma unroll
  for (int r = 0; r < 8; ++r) {
    const int m = m0 + r;
    float rg = sigm(ar[r] + bias[m]);
    float zg = sigm(az[r] + bias[Hc + m]);
    float ng = tanhf(anx[r] + bias[2 * Hc + m] + rg * anh[r]);
    float hv = (1.0f - zg) * ng + zg * bf2f(hold.us[r]);
    hnew.us[r] = f2bf(hv);
    if (outp) outp[(size_t)m * Nc] = hv;    // f32 result on last step
  }
  *(uint4*)(hnxt + off) = hnew.q;
}

// ---------------------------------------------------------------------------
// One GRU layer step for one wave of a 2-wave team: the wave computes its
// half of the gate-row tiles (htBase..htBase+7) for BOTH 16-sensor tiles of
// the team, reusing every A (weight) fragment against two B fragments.
// ---------------------------------------------------------------------------
template<bool XREG, int KTX>
__device__ __forceinline__ void gru_layer_pair(
    const Frag16* xf,                                   // XREG: xf[p*KTX+kt]
    const unsigned short* x0, const unsigned short* x1, // !XREG: LDS B buffers
    const unsigned short* __restrict__ pWx, const unsigned short* __restrict__ pWh,
    const float* __restrict__ bias,
    const unsigned short* hc0, unsigned short* hn0,
    const unsigned short* hc1, unsigned short* hn1,
    float* out0, float* out1, int lane, int htBase)
{
  for (int hti = 0; hti < 8; ++hti) {
    const int ht = htBase + hti;
    v8f ar0 = {}; v8f az0 = {}; v8f anx0 = {}; v8f anh0 = {};
    v8f ar1 = {}; v8f az1 = {}; v8f anx1 = {}; v8f anh1 = {};
    // input (x or lower-layer h) contribution: r, z, n_x
#pragma unroll
    for (int kt = 0; kt < KTX; ++kt) {
      Frag16 b0, b1;
      if (XREG) { b0 = xf[kt]; b1 = xf[KTX + kt]; }
      else      { b0 = load_bfrag(x0, kt, lane); b1 = load_bfrag(x1, kt, lane); }
      Frag16 a0 = load_afrag(pWx, ht,      KTX, kt, lane);
      Frag16 a1 = load_afrag(pWx, 16 + ht, KTX, kt, lane);
      Frag16 a2 = load_afrag(pWx, 32 + ht, KTX, kt, lane);
      wmma_bf16(ar0,  a0, b0);  wmma_bf16(ar1,  a0, b1);
      wmma_bf16(az0,  a1, b0);  wmma_bf16(az1,  a1, b1);
      wmma_bf16(anx0, a2, b0);  wmma_bf16(anx1, a2, b1);
    }
    // recurrent contribution: r, z, n_h
#pragma unroll 2
    for (int kt = 0; kt < 8; ++kt) {
      Frag16 b0 = load_bfrag(hc0, kt, lane);
      Frag16 b1 = load_bfrag(hc1, kt, lane);
      Frag16 a0 = load_afrag(pWh, ht,      8, kt, lane);
      Frag16 a1 = load_afrag(pWh, 16 + ht, 8, kt, lane);
      Frag16 a2 = load_afrag(pWh, 32 + ht, 8, kt, lane);
      wmma_bf16(ar0,  a0, b0);  wmma_bf16(ar1,  a0, b1);
      wmma_bf16(az0,  a1, b0);  wmma_bf16(az1,  a1, b1);
      wmma_bf16(anh0, a2, b0);  wmma_bf16(anh1, a2, b1);
    }
    gru_update(ar0, az0, anx0, anh0, bias, hc0, hn0, out0, ht, lane);
    gru_update(ar1, az1, anx1, anh1, bias, hc1, hn1, out1, ht, lane);
  }
}

// ---------------------------------------------------------------------------
// Main kernel. Block = 256 threads = 8 waves = 4 teams. Each team owns a
// 32-sensor group for one batch element across all T steps; state lives in
// team-private LDS (bf16, B-operand layout, ping-pong). Two split-barriers
// per step order layer0-write -> layer1-read and the ping-pong swap.
// ---------------------------------------------------------------------------
__global__ void __launch_bounds__(256) gru_main(
    const float* __restrict__ x,
    const unsigned short* __restrict__ pWx0, const unsigned short* __restrict__ pWh0,
    const unsigned short* __restrict__ pWx1, const unsigned short* __restrict__ pWh1,
    const float* __restrict__ b0, const float* __restrict__ b1,
    float* __restrict__ out)
{
  extern __shared__ unsigned short smem[];
  const int lane   = threadIdx.x & 31;
  const int wave   = threadIdx.x >> 5;
  const int team   = wave >> 1;
  const int htBase = (wave & 1) * 8;
  const int gid    = blockIdx.x * 4 + team;        // 0 .. 2047 sensor-pair groups
  const int tile0  = gid * 2;                      // first 16-col tile of group
  const int b      = tile0 >> 6;                   // N/16 == 64 tiles per batch
  const int n00    = (tile0 & 63) * 16;
  const int n      = lane & 15;
  const int khalf  = lane >> 4;
  const int tb     = team * 32768;                 // team region, ushorts
  // state offset: tb + layer*16384 + pingpong*8192 + ntile*4096

  // h0 = 0 : zero the whole state region once
  {
    uint4 z; z.x = z.y = z.z = z.w = 0u;
    for (int i = threadIdx.x; i < 16384; i += 256) ((uint4*)smem)[i] = z;
  }
  __syncthreads();

  float* o10 = out + ((size_t)b        * Hc) * Nc + n00 + n;   // layer0 final h
  float* o11 = o10 + 16;
  float* o20 = out + ((size_t)(Bc + b) * Hc) * Nc + n00 + n;   // layer1 final h
  float* o21 = o20 + 16;

  for (int t = 0; t < Tc; ++t) {
    const int cur = (t & 1) * 8192;
    const int nxt = 8192 - cur;
    unsigned short* h1c0 = smem + tb + cur;
    unsigned short* h1c1 = smem + tb + cur + 4096;
    unsigned short* h1n0 = smem + tb + nxt;
    unsigned short* h1n1 = smem + tb + nxt + 4096;
    unsigned short* h2c0 = smem + tb + 16384 + cur;
    unsigned short* h2c1 = smem + tb + 16384 + cur + 4096;
    unsigned short* h2n0 = smem + tb + 16384 + nxt;
    unsigned short* h2n1 = smem + tb + 16384 + nxt + 4096;

    // Stage x_t for both 16-sensor tiles as bf16 B-fragments in registers.
    Frag16 xf[4];                                   // [ntile*2 + kt]
    const float* xg = x + (((size_t)b * Tc + t) * Fc) * Nc + n00 + n;
#pragma unroll
    for (int j = 0; j < 16; ++j) {
      const size_t r0 = (size_t)(khalf * 16 + j) * Nc;
      const size_t r1 = (size_t)(32 + khalf * 16 + j) * Nc;
      xf[0].us[j] = f2bf(xg[r0]);
      xf[1].us[j] = f2bf(xg[r1]);
      xf[2].us[j] = f2bf(xg[r0 + 16]);
      xf[3].us[j] = f2bf(xg[r1 + 16]);
    }

    const bool last = (t == Tc - 1);
    gru_layer_pair<true, 2>(xf, nullptr, nullptr, pWx0, pWh0, b0,
                            h1c0, h1n0, h1c1, h1n1,
                            last ? o10 : nullptr, last ? o11 : nullptr,
                            lane, htBase);
    __syncthreads();   // both halves of h1_new ready before layer1 reads it
    gru_layer_pair<false, 8>(nullptr, h1n0, h1n1, pWx1, pWh1, b1,
                             h2c0, h2n0, h2c1, h2n1,
                             last ? o20 : nullptr, last ? o21 : nullptr,
                             lane, htBase);
    __syncthreads();   // step complete before ping-pong swap
  }
}

// ---------------------------------------------------------------------------
extern "C" void kernel_launch(void* const* d_in, const int* in_sizes, int n_in,
                              void* d_out, int out_size, void* d_ws, size_t ws_size,
                              hipStream_t stream) {
  (void)in_sizes; (void)n_in; (void)out_size; (void)ws_size;
  const float* x   = (const float*)d_in[0];
  const float* Wx0 = (const float*)d_in[1];
  const float* Wh0 = (const float*)d_in[2];
  const float* b0  = (const float*)d_in[3];
  const float* Wx1 = (const float*)d_in[4];
  const float* Wh1 = (const float*)d_in[5];
  const float* b1  = (const float*)d_in[6];
  float* out = (float*)d_out;

  // bf16 A-fragment weight buffers in workspace (1.25 MB total -> L2 resident)
  unsigned short* pWx0p = (unsigned short*)d_ws;
  unsigned short* pWh0p = pWx0p + (Gc / 16) * 2 * 512;
  unsigned short* pWx1p = pWh0p + (Gc / 16) * 8 * 512;
  unsigned short* pWh1p = pWx1p + (Gc / 16) * 8 * 512;

  const int tSmall = (Gc / 16) * 2 * 512;   // 49152
  const int tBig   = (Gc / 16) * 8 * 512;   // 196608
  pack_weights<<<(tSmall + 255) / 256, 256, 0, stream>>>(Wx0, pWx0p, 2);
  pack_weights<<<(tBig   + 255) / 256, 256, 0, stream>>>(Wh0, pWh0p, 8);
  pack_weights<<<(tBig   + 255) / 256, 256, 0, stream>>>(Wx1, pWx1p, 8);
  pack_weights<<<(tBig   + 255) / 256, 256, 0, stream>>>(Wh1, pWh1p, 8);

  (void)hipFuncSetAttribute(reinterpret_cast<const void*>(&gru_main),
                            hipFuncAttributeMaxDynamicSharedMemorySize, SMEM_BYTES);

  dim3 grid(512);                           // 2048 groups / 4 teams per block
  gru_main<<<grid, 256, SMEM_BYTES, stream>>>(
      x, pWx0p, pWh0p, pWx1p, pWh1p, b0, b1, out);
}
